// ButterflyBlock_31688268709984
// MI455X (gfx1250) — matrix operand order repaired and checked by reading, not compiled, not run on hardware
//
#include <hip/hip_runtime.h>
#include <hip/hip_bf16.h>

typedef _Float16 half8 __attribute__((ext_vector_type(8)));
typedef _Float16 v16h  __attribute__((ext_vector_type(16)));
typedef float    v8f   __attribute__((ext_vector_type(8)));

#define D_MODEL 1024
#define N_HEADS 16
#define HEAD_DIM 64
#define CHUNK 256
#define BATCH 4
#define SEQ 4096
#define TOK (BATCH * SEQ)   // 16384 tokens

__device__ __forceinline__ v16h cat16(half8 lo, half8 hi) {
  return __builtin_shufflevector(lo, hi, 0,1,2,3,4,5,6,7,8,9,10,11,12,13,14,15);
}

// ---------------------------------------------------------------------------
// GEMM: C[M,N] = A[M,K] @ B[K,N] + bias.  A is f32 (x) or f16 (attn out),
// B is f32 weights converted to f16 on the fly, accumulate f32 via WMMA,
// output f16 (workspace) or f32 (d_out).
// Block tile 128x128x32, 256 threads = 8 waves in 4(M) x 2(N); each wave
// computes 32x64 = 2x4 WMMA 16x16 tiles.
// ---------------------------------------------------------------------------
template<bool A_F32, bool OUT_F16>
__global__ __launch_bounds__(256)
void gemm_wmma(const void* __restrict__ Aptr, const float* __restrict__ Bptr,
               const float* __restrict__ bias, void* __restrict__ Cptr,
               int M, int K, int Nn) {
  constexpr int BM = 128, BN = 128, BK = 32, AST = 40, BST = 40;
  __shared__ alignas(16) _Float16 Ah[BM * AST];
  __shared__ alignas(16) _Float16 Bh[BN * BST];

  const int tid  = threadIdx.x;
  const int lane = tid & 31;
  const int wid  = tid >> 5;
  const int wm   = (wid & 3) * 32;   // wave M offset in tile
  const int wn   = (wid >> 2) * 64;  // wave N offset in tile
  const int m0   = blockIdx.y * BM;
  const int n0   = blockIdx.x * BN;
  const int cl   = lane & 15;
  const int hs   = lane >> 4;

  v8f acc[2][4];
  const v8f vzero = {0.f,0.f,0.f,0.f,0.f,0.f,0.f,0.f};
#pragma unroll
  for (int mt = 0; mt < 2; ++mt)
#pragma unroll
    for (int nt = 0; nt < 4; ++nt) acc[mt][nt] = vzero;

  const int ktiles = K / BK;
  for (int kt = 0; kt < ktiles; ++kt) {
    const int k0 = kt * BK;
    __syncthreads();
    // ---- stage A tile (convert to f16 if needed) ----
    if (A_F32) {
      const float* A = (const float*)Aptr;
#pragma unroll
      for (int i = 0; i < 4; ++i) {
        int idx = tid + i * 256;
        int r = idx >> 3, c = (idx & 7) << 2;
        float4 f = *(const float4*)(A + (size_t)(m0 + r) * K + k0 + c);
        Ah[r * AST + c + 0] = (_Float16)f.x;
        Ah[r * AST + c + 1] = (_Float16)f.y;
        Ah[r * AST + c + 2] = (_Float16)f.z;
        Ah[r * AST + c + 3] = (_Float16)f.w;
      }
    } else {
      const _Float16* A = (const _Float16*)Aptr;
#pragma unroll
      for (int i = 0; i < 2; ++i) {
        int idx = tid + i * 256;
        int r = idx >> 2, c = (idx & 3) << 3;
        *(half8*)&Ah[r * AST + c] =
            *(const half8*)(A + (size_t)(m0 + r) * K + k0 + c);
      }
    }
    // ---- stage B tile transposed: Bh[col][k] so B-fragments are contiguous
    // in the contraction dim (ISA 16-bit B layout: lane = column) ----
#pragma unroll
    for (int i = 0; i < 4; ++i) {
      int idx = tid + i * 256;
      int kr = idx >> 5, c = (idx & 31) << 2;
      float4 f = *(const float4*)(Bptr + (size_t)(k0 + kr) * Nn + n0 + c);
      Bh[(c + 0) * BST + kr] = (_Float16)f.x;
      Bh[(c + 1) * BST + kr] = (_Float16)f.y;
      Bh[(c + 2) * BST + kr] = (_Float16)f.z;
      Bh[(c + 3) * BST + kr] = (_Float16)f.w;
    }
    __syncthreads();

    // prefetch next K-tile into cache hierarchy (global_prefetch_b8)
    if (kt + 1 < ktiles) {
      if (A_F32)
        __builtin_prefetch((const float*)Aptr + (size_t)(m0 + (tid >> 1)) * K + k0 + BK, 0, 3);
      else
        __builtin_prefetch((const _Float16*)Aptr + (size_t)(m0 + (tid >> 1)) * K + k0 + BK, 0, 3);
      __builtin_prefetch(Bptr + (size_t)(k0 + BK + (tid >> 3)) * Nn + n0, 0, 3);
    }

    // ---- fragments + WMMA ----
    v16h afr[2], bfr[4];
#pragma unroll
    for (int mt = 0; mt < 2; ++mt) {
      int row = wm + mt * 16 + cl;
      half8 lo = *(const half8*)&Ah[row * AST + hs * 8];
      half8 hi = *(const half8*)&Ah[row * AST + hs * 8 + 16];
      afr[mt] = cat16(lo, hi);
    }
#pragma unroll
    for (int nt = 0; nt < 4; ++nt) {
      int col = wn + nt * 16 + cl;
      half8 lo = *(const half8*)&Bh[col * BST + hs * 8];
      half8 hi = *(const half8*)&Bh[col * BST + hs * 8 + 16];
      bfr[nt] = cat16(lo, hi);
    }
#pragma unroll
    for (int mt = 0; mt < 2; ++mt)
#pragma unroll
      for (int nt = 0; nt < 4; ++nt)
        acc[mt][nt] = __builtin_amdgcn_wmma_f32_16x16x32_f16(
            false, afr[mt], false, bfr[nt], (short)0, acc[mt][nt], false, false);
  }

  // ---- epilogue: bias add + store (C layout: row = r + 8*hs, col = cl) ----
#pragma unroll
  for (int nt = 0; nt < 4; ++nt) {
    int col = n0 + wn + nt * 16 + cl;
    float bval = bias[col];
#pragma unroll
    for (int mt = 0; mt < 2; ++mt) {
#pragma unroll
      for (int r = 0; r < 8; ++r) {
        int row = m0 + wm + mt * 16 + r + 8 * hs;
        float v = acc[mt][nt][r] + bval;
        if (OUT_F16)
          ((_Float16*)Cptr)[(size_t)row * Nn + col] = (_Float16)v;
        else
          ((float*)Cptr)[(size_t)row * Nn + col] = v;
      }
    }
  }
}

// ---------------------------------------------------------------------------
// Flash attention over one butterfly pair-block (512 tokens = chunk cA ++ cB).
// Grid: (8 q-tiles of 64, 16 heads, 32 = batch*pairs). 128 threads = 4 waves,
// each wave owns 16 query rows; loops over 8 key tiles of 64 keys.
// ---------------------------------------------------------------------------
__global__ __launch_bounds__(128)
void attn_flash(const _Float16* __restrict__ Qh, const _Float16* __restrict__ Kh,
                const _Float16* __restrict__ Vh, _Float16* __restrict__ Oh,
                const int* __restrict__ layer_bit) {
  constexpr int PST = 72, VST = 72;
  __shared__ alignas(16) _Float16 VT[64 * VST];       // V tile transposed [dim][key]
  __shared__ alignas(16) _Float16 Pl[4 * 16 * PST];   // per-wave P tiles

  const int tid = threadIdx.x, lane = tid & 31, w = tid >> 5;
  const int cl = lane & 15, hs = lane >> 4;
  const int head = blockIdx.y;
  const int z = blockIdx.z, b = z >> 3, pi = z & 7;
  const int lb = *layer_bit;
  const int cA = ((pi >> lb) << (lb + 1)) | (pi & ((1 << lb) - 1));
  const int cB = cA | (1 << lb);
  const int base = b * SEQ;
  auto tok = [&](int i) {
    return base + ((i < CHUNK) ? (cA * CHUNK + i) : (cB * CHUNK + (i - CHUNK)));
  };

  const int q0 = blockIdx.x * 64 + w * 16;           // wave's first local query
  const size_t hoff = (size_t)head * HEAD_DIM;

  // Q A-fragments (row = cl, K split by hs per ISA 16-bit A layout)
  v16h qa[2];
  {
    int qt = tok(q0 + cl);
    const _Float16* qp = Qh + (size_t)qt * D_MODEL + hoff;
#pragma unroll
    for (int kk = 0; kk < 2; ++kk) {
      half8 lo = *(const half8*)(qp + kk * 32 + hs * 8);
      half8 hi = *(const half8*)(qp + kk * 32 + hs * 8 + 16);
      qa[kk] = cat16(lo, hi);
    }
  }

  const v8f vzero = {0.f,0.f,0.f,0.f,0.f,0.f,0.f,0.f};
  v8f o[4];
#pragma unroll
  for (int nt = 0; nt < 4; ++nt) o[nt] = vzero;
  float mrow[8], lrow[8];
#pragma unroll
  for (int r = 0; r < 8; ++r) { mrow[r] = -1e30f; lrow[r] = 0.f; }
  const float scale = 0.125f;  // 1/sqrt(64)

  for (int kt = 0; kt < 8; ++kt) {
    const int kbase = kt * 64;

    // ---- stage V^T tile cooperatively ----
    __syncthreads();
    {
      int kr = tid >> 1, seg = tid & 1;
      int vt = tok(kbase + kr);
      const _Float16* vp = Vh + (size_t)vt * D_MODEL + hoff + seg * 32;
#pragma unroll
      for (int j = 0; j < 4; ++j) {
        half8 v = *(const half8*)(vp + j * 8);
#pragma unroll
        for (int e = 0; e < 8; ++e)
          VT[(seg * 32 + j * 8 + e) * VST + kr] = v[e];
      }
    }
    __syncthreads();

    // ---- S = (Q K^T) * scale : K B-fragments straight from global ----
    v8f s[4];
#pragma unroll
    for (int nt = 0; nt < 4; ++nt) s[nt] = vzero;
#pragma unroll
    for (int nt = 0; nt < 4; ++nt) {
      int kv = tok(kbase + nt * 16 + cl);
      const _Float16* kp = Kh + (size_t)kv * D_MODEL + hoff;
#pragma unroll
      for (int kk = 0; kk < 2; ++kk) {
        half8 lo = *(const half8*)(kp + kk * 32 + hs * 8);
        half8 hi = *(const half8*)(kp + kk * 32 + hs * 8 + 16);
        v16h bf = cat16(lo, hi);
        s[nt] = __builtin_amdgcn_wmma_f32_16x16x32_f16(
            false, qa[kk], false, bf, (short)0, s[nt], false, false);
      }
    }

    // ---- online softmax (row = r + 8*hs spans 16 lanes of one half) ----
    float rmax[8];
#pragma unroll
    for (int r = 0; r < 8; ++r) rmax[r] = mrow[r];
#pragma unroll
    for (int nt = 0; nt < 4; ++nt)
#pragma unroll
      for (int r = 0; r < 8; ++r) {
        s[nt][r] *= scale;
        rmax[r] = fmaxf(rmax[r], s[nt][r]);
      }
#pragma unroll
    for (int r = 0; r < 8; ++r)
#pragma unroll
      for (int msk = 8; msk >= 1; msk >>= 1)
        rmax[r] = fmaxf(rmax[r], __shfl_xor(rmax[r], msk, 32));

    float alpha[8], rsum[8];
#pragma unroll
    for (int r = 0; r < 8; ++r) {
      alpha[r] = __expf(mrow[r] - rmax[r]);
      mrow[r] = rmax[r];
      rsum[r] = 0.f;
    }
#pragma unroll
    for (int nt = 0; nt < 4; ++nt)
#pragma unroll
      for (int r = 0; r < 8; ++r) {
        float p = __expf(s[nt][r] - mrow[r]);
        s[nt][r] = p;
        rsum[r] += p;
      }
#pragma unroll
    for (int r = 0; r < 8; ++r)
#pragma unroll
      for (int msk = 8; msk >= 1; msk >>= 1)
        rsum[r] += __shfl_xor(rsum[r], msk, 32);
#pragma unroll
    for (int r = 0; r < 8; ++r) lrow[r] = lrow[r] * alpha[r] + rsum[r];
#pragma unroll
    for (int nt = 0; nt < 4; ++nt)
#pragma unroll
      for (int r = 0; r < 8; ++r) o[nt][r] *= alpha[r];

    // ---- P (C-layout) -> LDS -> A-fragments ----
    _Float16* pw = &Pl[w * 16 * PST];
#pragma unroll
    for (int nt = 0; nt < 4; ++nt)
#pragma unroll
      for (int r = 0; r < 8; ++r)
        pw[(r + 8 * hs) * PST + nt * 16 + cl] = (_Float16)s[nt][r];
    asm volatile("s_wait_dscnt 0" ::: "memory");

    v16h pa[2];
#pragma unroll
    for (int kk = 0; kk < 2; ++kk) {
      half8 lo = *(const half8*)&pw[cl * PST + kk * 32 + hs * 8];
      half8 hi = *(const half8*)&pw[cl * PST + kk * 32 + hs * 8 + 16];
      pa[kk] = cat16(lo, hi);
    }

    // ---- O += P @ V : V^T B-fragments from LDS (contiguous in key dim) ----
#pragma unroll
    for (int nt = 0; nt < 4; ++nt) {
      int dim = nt * 16 + cl;
#pragma unroll
      for (int kk = 0; kk < 2; ++kk) {
        half8 lo = *(const half8*)&VT[dim * VST + kk * 32 + hs * 8];
        half8 hi = *(const half8*)&VT[dim * VST + kk * 32 + hs * 8 + 16];
        v16h vb = cat16(lo, hi);
        o[nt] = __builtin_amdgcn_wmma_f32_16x16x32_f16(
            false, pa[kk], false, vb, (short)0, o[nt], false, false);
      }
    }
  }

  // ---- normalize + store attention output (f16) ----
  float rinv[8];
#pragma unroll
  for (int r = 0; r < 8; ++r) rinv[r] = 1.0f / fmaxf(lrow[r], 1e-30f);
#pragma unroll
  for (int r = 0; r < 8; ++r) {
    int qt = tok(q0 + r + 8 * hs);
    _Float16* op = Oh + (size_t)qt * D_MODEL + hoff;
#pragma unroll
    for (int nt = 0; nt < 4; ++nt)
      op[nt * 16 + cl] = (_Float16)(o[nt][r] * rinv[r]);
  }
}

// ---------------------------------------------------------------------------
extern "C" void kernel_launch(void* const* d_in, const int* in_sizes, int n_in,
                              void* d_out, int out_size, void* d_ws, size_t ws_size,
                              hipStream_t stream) {
  (void)in_sizes; (void)n_in; (void)out_size; (void)ws_size;
  const float* x  = (const float*)d_in[0];
  const float* Wq = (const float*)d_in[1];
  const float* bq = (const float*)d_in[2];
  const float* Wk = (const float*)d_in[3];
  const float* bk = (const float*)d_in[4];
  const float* Wv = (const float*)d_in[5];
  const float* bv = (const float*)d_in[6];
  const float* Wo = (const float*)d_in[7];
  const float* bo = (const float*)d_in[8];
  const int*   lb = (const int*)d_in[9];

  const size_t matBytes = (size_t)TOK * D_MODEL * sizeof(_Float16);
  _Float16* Qh = (_Float16*)d_ws;
  _Float16* Kh = (_Float16*)((char*)d_ws + 1 * matBytes);
  _Float16* Vh = (_Float16*)((char*)d_ws + 2 * matBytes);
  _Float16* Oh = (_Float16*)((char*)d_ws + 3 * matBytes);

  dim3 ggrid(D_MODEL / 128, TOK / 128);
  gemm_wmma<true, true><<<ggrid, 256, 0, stream>>>(x, Wq, bq, Qh, TOK, D_MODEL, D_MODEL);
  gemm_wmma<true, true><<<ggrid, 256, 0, stream>>>(x, Wk, bk, Kh, TOK, D_MODEL, D_MODEL);
  gemm_wmma<true, true><<<ggrid, 256, 0, stream>>>(x, Wv, bv, Vh, TOK, D_MODEL, D_MODEL);

  attn_flash<<<dim3(8, N_HEADS, 32), 128, 0, stream>>>(Qh, Kh, Vh, Oh, lb);

  gemm_wmma<false, false><<<ggrid, 256, 0, stream>>>(Oh, Wo, bo, (float*)d_out,
                                                     TOK, D_MODEL, D_MODEL);
}
